// TritonFlashCausalAttention_34170759807483
// MI455X (gfx1250) — compile-verified, matrix operand order checked by reading, not run on hardware
//
#include <hip/hip_runtime.h>

#define S_LEN   2048
#define D_MODEL 2048
#define N_HEADS 16
#define HEAD_DIM 128
#define BATCH   2
#define M_TOTAL (BATCH * S_LEN) /* 4096 */

typedef __attribute__((ext_vector_type(16))) __bf16 v16bf;
typedef __attribute__((ext_vector_type(8)))  __bf16 v8bf;
typedef __attribute__((ext_vector_type(4)))  __bf16 v4bf;
typedef __attribute__((ext_vector_type(8)))  float  v8f;

// ---------------------------------------------------------------------------
// CDNA5 async global->LDS copy (ASYNCcnt-tracked, no VGPR data round trip).
// LDS destination operand is addr[31:0] of the generic pointer (LDS aperture).
// ---------------------------------------------------------------------------
__device__ __forceinline__ void async_copy_b64(void* lds, const void* gptr) {
  asm volatile("global_load_async_to_lds_b64 %0, %1, off"
               :: "v"((unsigned)(unsigned long long)lds),
                  "v"((unsigned long long)gptr)
               : "memory");
}
__device__ __forceinline__ void wait_async0() {
  asm volatile("s_wait_asynccnt 0" ::: "memory");
}

// ---------------------------------------------------------------------------
// Fragment loader for 16-bit A/B operands of V_WMMA_F32_16X16X32_BF16.
// A (16x32, MxK): lane L holds row M=L&15, K={kh..kh+7, 16+kh..16+kh+7},
// kh = 8*(L>>4). B mirrors with lane -> column N, so X*W^T loads both operands
// from row-major storage with this helper. Two ds_load_b128 per lane.
// ---------------------------------------------------------------------------
__device__ __forceinline__ v16bf load_frag_bf16(const __bf16* p, int ld) {
  const int lane = (int)(threadIdx.x & 31u);
  const __bf16* r = p + (lane & 15) * ld + ((lane >> 4) << 3);
  v8bf lo = *(const v8bf*)(r);
  v8bf hi = *(const v8bf*)(r + 16);
  return __builtin_shufflevector(lo, hi, 0,1,2,3,4,5,6,7,8,9,10,11,12,13,14,15);
}

// wave32 xor-lane reductions via ds_bpermute (no LDS memory touched)
__device__ __forceinline__ float lane_xor_max(float v, int m) {
  int idx = (((int)(threadIdx.x & 31u)) ^ m) << 2;
  return fmaxf(v, __int_as_float(__builtin_amdgcn_ds_bpermute(idx, __float_as_int(v))));
}
__device__ __forceinline__ float lane_xor_add(float v, int m) {
  int idx = (((int)(threadIdx.x & 31u)) ^ m) << 2;
  return v + __int_as_float(__builtin_amdgcn_ds_bpermute(idx, __float_as_int(v)));
}

// ---------------------------------------------------------------------------
// f32 -> bf16 conversion (grid-stride)
// ---------------------------------------------------------------------------
__global__ __launch_bounds__(256) void mha_cvt_bf16(const float* __restrict__ in,
                                                    __bf16* __restrict__ out, int n) {
  int i = (int)(blockIdx.x * 256 + threadIdx.x);
  int step = (int)(gridDim.x * 256);
  for (; i < n; i += step) out[i] = (__bf16)in[i];
}

// ---------------------------------------------------------------------------
// C[M,N] = Xbf16[M,K] * W[N,K]^T + bias.
// WG tile 256x128, K-step 32, double-buffered async LDS fill.
// 8 waves: wm in 0..3 (64-row group), wn in 0..1 (64-col group)
// -> each wave owns a 64x64 = 4x4 WMMA accumulator block
//    (16 WMMAs per 8 fragment loads per K-step).
// MODE 0: write f32 row-major (out projection).
// MODE 1: split columns [3,H,Dh] into Q/K/V bf16 buffers (qkv projection).
// ---------------------------------------------------------------------------
template<int MODE>
__global__ __launch_bounds__(256) void mha_gemm_bf16(
    const __bf16* __restrict__ X, const __bf16* __restrict__ W,
    const float* __restrict__ bias,
    float*  __restrict__ outF,
    __bf16* __restrict__ outQ, __bf16* __restrict__ outK, __bf16* __restrict__ outV,
    int M, int N, int K)
{
  constexpr int LDT = 40; // padded LDS stride (keeps 16B fragment alignment)
  __shared__ __align__(16) __bf16 Xs[2][256 * LDT];
  __shared__ __align__(16) __bf16 Wsh[2][128 * LDT];

  const int tid  = (int)threadIdx.x;
  const int lane = tid & 31;
  const int w    = tid >> 5;
  const int wm   = w & 3;
  const int wn   = w >> 2;
  const int m0   = (int)blockIdx.y * 256;
  const int n0   = (int)blockIdx.x * 128;

  // Issue async fill of one K-tile (256x32 X rows + 128x32 W rows = 3072 b64).
  auto issue_tile = [&](int k0, int buf) {
    #pragma unroll
    for (int i = 0; i < 12; ++i) {
      int c = tid + i * 256;
      if (c < 2048) {
        int row = c >> 3, col = (c & 7) << 2;
        async_copy_b64(&Xs[buf][row * LDT + col],
                       &X[(size_t)(m0 + row) * K + k0 + col]);
      } else {
        int c2 = c - 2048;
        int row = c2 >> 3, col = (c2 & 7) << 2;
        async_copy_b64(&Wsh[buf][row * LDT + col],
                       &W[(size_t)(n0 + row) * K + k0 + col]);
      }
    }
  };

  v8f acc[4][4] = {};

  issue_tile(0, 0);
  wait_async0();
  __syncthreads();

  int cur = 0;
  for (int k0 = 0; k0 < K; k0 += 32) {
    if (k0 + 32 < K) issue_tile(k0 + 32, cur ^ 1);   // overlap copy with math

    v16bf bfr[4];
    #pragma unroll
    for (int ni = 0; ni < 4; ++ni)
      bfr[ni] = load_frag_bf16(&Wsh[cur][(wn * 64 + ni * 16) * LDT], LDT);
    #pragma unroll
    for (int mi = 0; mi < 4; ++mi) {
      v16bf a = load_frag_bf16(&Xs[cur][(wm * 64 + mi * 16) * LDT], LDT);
      #pragma unroll
      for (int ni = 0; ni < 4; ++ni)
        acc[mi][ni] = __builtin_amdgcn_wmma_f32_16x16x32_bf16(
            false, a, false, bfr[ni], (short)0, acc[mi][ni], false, false);
    }

    wait_async0();       // next buffer resident in LDS
    __syncthreads();     // everyone done reading cur before it is refilled
    cur ^= 1;
  }

  // Epilogue. C layout: lane -> n = lane&15; element j -> m = j + 8*(lane>>4).
  const int nloc = lane & 15;
  const int mh   = (lane >> 4) << 3;
  #pragma unroll
  for (int mi = 0; mi < 4; ++mi) {
    #pragma unroll
    for (int ni = 0; ni < 4; ++ni) {
      #pragma unroll
      for (int j = 0; j < 8; ++j) {
        int row = m0 + wm * 64 + mi * 16 + mh + j;
        int col = n0 + wn * 64 + ni * 16 + nloc;
        float v = acc[mi][ni][j] + bias[col];
        if (MODE == 0) {
          outF[(size_t)row * N + col] = v;
        } else {
          int c  = col >> 11;          // 0:q 1:k 2:v  (N = 3*2048)
          int cc = col & 2047;         // h*128 + dh
          __bf16* dst = (c == 0) ? outQ : ((c == 1) ? outK : outV);
          dst[(size_t)row * D_MODEL + cc] = (__bf16)v;
        }
      }
    }
  }
}

// ---------------------------------------------------------------------------
// Flash attention, causal + additive mask. One block per (b, h, 64 query rows).
// 8 waves: wm in 0..3 -> 16-row group, wn in 0..1 -> 64-wide half of Dh=128.
// Key blocks of 64; online softmax in f32; P re-laid out via LDS for P*V.
// Q/K tiles fill via async global->LDS; V is VALU-transposed into [dh][key].
// ---------------------------------------------------------------------------
__global__ __launch_bounds__(256) void mha_flash_attn(
    const __bf16* __restrict__ Qg, const __bf16* __restrict__ Kg,
    const __bf16* __restrict__ Vg, const float* __restrict__ maskg,
    __bf16* __restrict__ ctx)
{
  constexpr int LQ = 136, LK = 136, LV = 72, LP = 72;
  __shared__ __align__(16) __bf16 Qs[64 * LQ];
  __shared__ __align__(16) __bf16 Ks[64 * LK];
  __shared__ __align__(16) __bf16 Vst[128 * LV];     // V transposed: [dh][key]
  __shared__ __align__(16) __bf16 Ps[4][16 * LP];    // per-row-group P tile

  const int qb   = (int)blockIdx.x;       // 64-row query block
  const int h    = (int)blockIdx.y;
  const int b    = (int)blockIdx.z;
  const int tid  = (int)threadIdx.x;
  const int lane = tid & 31;
  const int w    = tid >> 5;
  const int wm   = w & 3;
  const int wn   = w >> 2;
  const int nloc = lane & 15;
  const int mh   = (lane >> 4) << 3;
  const size_t rowbase = (size_t)b * S_LEN + (size_t)qb * 64;
  const float scale = 0.08838834764831845f;          // 1/sqrt(128)

  // Async-load Q tile 64 x 128 (2048 b64 chunks); waited in first iteration.
  #pragma unroll
  for (int i = 0; i < 8; ++i) {
    int c = tid + i * 256, row = c >> 5, col = (c & 31) << 2;
    async_copy_b64(&Qs[row * LQ + col],
                   &Qg[(rowbase + row) * D_MODEL + h * HEAD_DIM + col]);
  }

  float mrow[8], lrow[8];
  #pragma unroll
  for (int j = 0; j < 8; ++j) { mrow[j] = -__builtin_inff(); lrow[j] = 0.0f; }
  v8f acc[4] = {};                                  // 16 x 64 ctx accumulator

  for (int kb = 0; kb <= qb; ++kb) {
    const int t0 = kb * 64;
    __syncthreads();                                // prior reads of Ks/Vst done
    #pragma unroll
    for (int i = 0; i < 8; ++i) {                   // K tile row-major (async)
      int c = tid + i * 256, row = c >> 5, col = (c & 31) << 2;
      async_copy_b64(&Ks[row * LK + col],
                     &Kg[((size_t)b * S_LEN + t0 + row) * D_MODEL + h * HEAD_DIM + col]);
    }
    #pragma unroll
    for (int i = 0; i < 8; ++i) {                   // V tile transposed (VALU)
      int c = tid + i * 256, row = c >> 5, col = (c & 31) << 2;
      v4bf tv = *(const v4bf*)&Vg[((size_t)b * S_LEN + t0 + row) * D_MODEL + h * HEAD_DIM + col];
      #pragma unroll
      for (int jj = 0; jj < 4; ++jj) Vst[(col + jj) * LV + row] = tv[jj];
    }
    wait_async0();                                  // Q (first iter) + K resident
    __syncthreads();

    // S = Q * K^T  (16 x 64 per wave, Dh reduced in 4 K=32 steps)
    v8f sacc[4] = {};
    #pragma unroll
    for (int kk = 0; kk < 4; ++kk) {
      v16bf aq = load_frag_bf16(&Qs[(wm * 16) * LQ + kk * 32], LQ);
      #pragma unroll
      for (int ni = 0; ni < 4; ++ni) {
        v16bf bk = load_frag_bf16(&Ks[(ni * 16) * LK + kk * 32], LK);
        sacc[ni] = __builtin_amdgcn_wmma_f32_16x16x32_bf16(
            false, aq, false, bk, (short)0, sacc[ni], false, false);
      }
    }

    // scale + mask + causal, then online softmax (rows live in 16-lane halves)
    #pragma unroll
    for (int j = 0; j < 8; ++j) {
      const int r = qb * 64 + wm * 16 + mh + j;
      float mx = -3.0e38f;
      #pragma unroll
      for (int ni = 0; ni < 4; ++ni) {
        const int t = t0 + ni * 16 + nloc;
        float s = sacc[ni][j] * scale + maskg[((size_t)h * S_LEN + r) * S_LEN + t];
        if (t > r) s = -3.0e38f;
        sacc[ni][j] = s;
        mx = fmaxf(mx, s);
      }
      mx = lane_xor_max(mx, 1); mx = lane_xor_max(mx, 2);
      mx = lane_xor_max(mx, 4); mx = lane_xor_max(mx, 8);
      const float mnew = fmaxf(mrow[j], mx);
      const float corr = __expf(mrow[j] - mnew);
      mrow[j] = mnew;
      float rs = 0.0f;
      #pragma unroll
      for (int ni = 0; ni < 4; ++ni) {
        float p = __expf(sacc[ni][j] - mnew);
        sacc[ni][j] = p;
        rs += p;
      }
      rs = lane_xor_add(rs, 1); rs = lane_xor_add(rs, 2);
      rs = lane_xor_add(rs, 4); rs = lane_xor_add(rs, 8);
      lrow[j] = lrow[j] * corr + rs;
      #pragma unroll
      for (int vi = 0; vi < 4; ++vi) acc[vi][j] *= corr;
    }

    // Spill P (C layout) to LDS, reload as A fragments for P*V.
    #pragma unroll
    for (int j = 0; j < 8; ++j)
      #pragma unroll
      for (int ni = 0; ni < 4; ++ni)
        Ps[wm][(mh + j) * LP + ni * 16 + nloc] = (__bf16)sacc[ni][j];
    asm volatile("s_wait_dscnt 0" ::: "memory");    // own-wave LDS RAW fence

    #pragma unroll
    for (int kk = 0; kk < 2; ++kk) {
      v16bf ap = load_frag_bf16(&Ps[wm][kk * 32], LP);
      #pragma unroll
      for (int vi = 0; vi < 4; ++vi) {
        v16bf bv = load_frag_bf16(&Vst[(wn * 64 + vi * 16) * LV + kk * 32], LV);
        acc[vi] = __builtin_amdgcn_wmma_f32_16x16x32_bf16(
            false, ap, false, bv, (short)0, acc[vi], false, false);
      }
    }
  }

  // Normalize and emit ctx (bf16) at [b, s, h*Dh + col].
  #pragma unroll
  for (int j = 0; j < 8; ++j) {
    const float inv = 1.0f / lrow[j];
    const size_t row = rowbase + wm * 16 + mh + j;
    #pragma unroll
    for (int vi = 0; vi < 4; ++vi) {
      int col = h * HEAD_DIM + wn * 64 + vi * 16 + nloc;
      ctx[row * D_MODEL + col] = (__bf16)(acc[vi][j] * inv);
    }
  }
}

// ---------------------------------------------------------------------------
// Host-side launch. Workspace layout (bytes):
//   xb     @         0 : 4096*2048 bf16   (16 MB)
//   wqkvb  @  16777216 : 6144*2048 bf16   (24 MB)
//   woutb  @  41943040 : 2048*2048 bf16   ( 8 MB)
//   qb     @  50331648 : 4096*2048 bf16   (16 MB)
//   kb     @  67108864 : 4096*2048 bf16   (16 MB)
//   vb     @  83886080 : 4096*2048 bf16   (16 MB)
//   ctxb   @ 100663296 : 4096*2048 bf16   (16 MB)   total 112 MB
// ---------------------------------------------------------------------------
extern "C" void kernel_launch(void* const* d_in, const int* in_sizes, int n_in,
                              void* d_out, int out_size, void* d_ws, size_t ws_size,
                              hipStream_t stream) {
  const float* x     = (const float*)d_in[0];
  const float* amask = (const float*)d_in[1];
  const float* Wqkv  = (const float*)d_in[2];
  const float* bqkv  = (const float*)d_in[3];
  const float* Wout  = (const float*)d_in[4];
  const float* bout  = (const float*)d_in[5];
  float* out = (float*)d_out;

  char* ws = (char*)d_ws;
  __bf16* xb    = (__bf16*)(ws + 0);
  __bf16* wqkvb = (__bf16*)(ws + 16777216);
  __bf16* woutb = (__bf16*)(ws + 41943040);
  __bf16* qb    = (__bf16*)(ws + 50331648);
  __bf16* kb    = (__bf16*)(ws + 67108864);
  __bf16* vb    = (__bf16*)(ws + 83886080);
  __bf16* ctxb  = (__bf16*)(ws + 100663296);

  mha_cvt_bf16<<<4096, 256, 0, stream>>>(x,    xb,    M_TOTAL * D_MODEL);
  mha_cvt_bf16<<<4096, 256, 0, stream>>>(Wqkv, wqkvb, 3 * D_MODEL * D_MODEL);
  mha_cvt_bf16<<<4096, 256, 0, stream>>>(Wout, woutb, D_MODEL * D_MODEL);

  // qkv = x @ Wqkv^T + bqkv  -> split into q/k/v (bf16)
  mha_gemm_bf16<1><<<dim3(3 * D_MODEL / 128, M_TOTAL / 256), 256, 0, stream>>>(
      xb, wqkvb, bqkv, nullptr, qb, kb, vb, M_TOTAL, 3 * D_MODEL, D_MODEL);

  // flash attention -> ctx (bf16)
  mha_flash_attn<<<dim3(S_LEN / 64, N_HEADS, BATCH), 256, 0, stream>>>(
      qb, kb, vb, amask, ctxb);

  // out = ctx @ Wout^T + bout  (f32)
  mha_gemm_bf16<0><<<dim3(D_MODEL / 128, M_TOTAL / 256), 256, 0, stream>>>(
      ctxb, woutb, bout, out, nullptr, nullptr, nullptr, M_TOTAL, D_MODEL, D_MODEL);
}
